// AnotherGRUCell_43181601194448
// MI455X (gfx1250) — matrix operand order repaired and checked by reading, not compiled, not run on hardware
//
#include <hip/hip_runtime.h>
#include <hip/hip_bf16.h>

// ---------------------------------------------------------------------------
// GRU cell, MI455X (gfx1250).
//   h_t = u*h + (1-u)*tanh(xh + r*(h@Wh3) + bh)
//   r = sigmoid(xr+hr+br), u = sigmoid(xu+hu+bu)
// bf16 WMMA (16x16x32), f32 accumulate.
//  - Weights packed bf16 with gates interleaved per (unit-tile, k-tile) so a
//    single pointer + 24-bit immediate offsets cover the whole K walk.
//  - Each wave computes 2 M-tiles (32 rows) x 1 N-tile: B fragments reused
//    twice in registers -> 12 WMMAs per 20 b128 loads.
//  - Two named fragment sets (fa/fb), 4-tile unrolled software pipeline.
// ---------------------------------------------------------------------------

typedef __bf16 v16bf __attribute__((ext_vector_type(16)));
typedef float  v8f   __attribute__((ext_vector_type(8)));

#define BATCH   8192
#define KDIM    2048          // IN_DIM == UNITS == 2048
#define UNITS   2048
#define NCOL    (3 * UNITS)   // 6144
#define NT_N    (NCOL / 16)   // 384 column tiles (16 wide)
#define NT_K    (KDIM / 32)   // 64  K tiles (32 deep)

__device__ __forceinline__ __bf16 f2bf(float f) {
    union { float f; unsigned u; } c; c.f = f;
    unsigned r = (c.u + 0x7FFFu + ((c.u >> 16) & 1u)) >> 16;  // RNE
    union { unsigned short s; __bf16 b; } o; o.s = (unsigned short)r;
    return o.b;
}

// ---- pass 1a: f32 -> bf16, row major (x_t, h_tm1 => A operands) -----------
__global__ __launch_bounds__(256)
void conv_a_bf16(const float* __restrict__ src, __bf16* __restrict__ dst, int n) {
    int i = (blockIdx.x * blockDim.x + threadIdx.x) * 8;
    if (i >= n) return;
    float4 a = *(const float4*)(src + i);
    float4 b = *(const float4*)(src + i + 4);
    __align__(16) __bf16 o[8] = { f2bf(a.x), f2bf(a.y), f2bf(a.z), f2bf(a.w),
                                  f2bf(b.x), f2bf(b.y), f2bf(b.z), f2bf(b.w) };
    *(uint4*)(dst + i) = *(const uint4*)o;
}

// ---- pass 1b: pack weights [K=2048][N=6144] into WMMA B-fragment tiles ----
// Logical column n = gate*2048 + ntu*16 + (lane&15), gate in {0,1,2}.
// Tile (ntu, kt, gate) = 32x16 bf16; lane l holds column (l&15),
// K = kt*32 + (l>>4)*16 .. +15, contiguous (16 bf16 = 32B per lane).
// dst offset: (((ntu*64 + kt)*3 + gate)*32 + l) * 16 elements  (gate-interleaved)
__global__ __launch_bounds__(256)
void pack_w_bf16(const float* __restrict__ W, __bf16* __restrict__ dst) {
    int j    = blockIdx.x * blockDim.x + threadIdx.x;   // one job per tile-lane
    int lane = j & 31;
    int tile = j >> 5;                                  // 0 .. NT_N*NT_K-1
    if (tile >= NT_N * NT_K) return;
    int kt   = tile & (NT_K - 1);
    int nt   = tile >> 6;          // 0..383 logical column tile
    int gate = nt >> 7;            // 0..2
    int ntu  = nt & 127;           // unit column tile
    int n    = nt * 16 + (lane & 15);
    int k0   = kt * 32 + (lane >> 4) * 16;
    __align__(16) __bf16 o[16];
#pragma unroll
    for (int i = 0; i < 16; ++i)
        o[i] = f2bf(W[(size_t)(k0 + i) * NCOL + n]);
    size_t dtile = ((size_t)ntu * NT_K + kt) * 3 + gate;
    uint4* d = (uint4*)(dst + (dtile * 32 + lane) * 16);
    d[0] = *(const uint4*)&o[0];
    d[1] = *(const uint4*)&o[8];
}

// A fragment: one contiguous 32B run.  B fragment: one contiguous 32B run.
__device__ __forceinline__ v16bf ldA(const __bf16* p, int e) {
    union { v16bf v; uint4 u[2]; } f;
    f.u[0] = *(const uint4*)(p + e);
    f.u[1] = *(const uint4*)(p + e + 8);
    return f.v;
}
__device__ __forceinline__ v16bf ldB(const __bf16* p, int e) {
    union { v16bf v; uint4 u[2]; } f;
    f.u[0] = *(const uint4*)(p + e);
    f.u[1] = *(const uint4*)(p + e + 8);
    return f.v;
}

__device__ __forceinline__ v8f wmma_bf16(v16bf a, v16bf b, v8f c) {
    return __builtin_amdgcn_wmma_f32_16x16x32_bf16(
        /*neg_a*/false, a, /*neg_b*/false, b,
        /*c_mod*/(short)0, c, /*reuse_a*/false, /*reuse_b*/false);
}

struct Ptrs { const __bf16 *xA, *hA, *wi, *wh; };

// One K-tile of fragments: 2 M-tiles of A (x,h) + 6 gate-interleaved B tiles.
struct Frag10 { v16bf ax0, ax1, ah0, ah1, br, bu, bc, cr, cu, ch; };

// t is a small compile-time-constant tile index relative to current pointers.
__device__ __forceinline__ Frag10 load_all(const Ptrs& p, int t) {
    Frag10 f;
    const int ka = t * 32;            // A elements per tile
    const int kw = t * 3 * 512;       // packed-W elements per tile
    f.ax0 = ldA(p.xA, ka);
    f.ax1 = ldA(p.xA, ka + 16 * KDIM);   // +16 rows (65536 B immediate)
    f.ah0 = ldA(p.hA, ka);
    f.ah1 = ldA(p.hA, ka + 16 * KDIM);
    f.br  = ldB(p.wi, kw);
    f.bu  = ldB(p.wi, kw + 512);
    f.bc  = ldB(p.wi, kw + 1024);
    f.cr  = ldB(p.wh, kw);
    f.cu  = ldB(p.wh, kw + 512);
    f.ch  = ldB(p.wh, kw + 1024);
    return f;
}

// 12 WMMAs; same-accumulator pairs are 6 apart (hazard-free spacing).
__device__ __forceinline__ void mma_all(const Frag10& f,
        v8f& r0, v8f& u0, v8f& c0, v8f& h0,
        v8f& r1, v8f& u1, v8f& c1, v8f& h1) {
    r0 = wmma_bf16(f.ax0, f.br, r0);
    u0 = wmma_bf16(f.ax0, f.bu, u0);
    c0 = wmma_bf16(f.ax0, f.bc, c0);
    r1 = wmma_bf16(f.ax1, f.br, r1);
    u1 = wmma_bf16(f.ax1, f.bu, u1);
    c1 = wmma_bf16(f.ax1, f.bc, c1);
    r0 = wmma_bf16(f.ah0, f.cr, r0);
    u0 = wmma_bf16(f.ah0, f.cu, u0);
    h0 = wmma_bf16(f.ah0, f.ch, h0);
    r1 = wmma_bf16(f.ah1, f.cr, r1);
    u1 = wmma_bf16(f.ah1, f.cu, u1);
    h1 = wmma_bf16(f.ah1, f.ch, h1);
}

// ---- pass 2: fused dual-GEMM + GRU epilogue -------------------------------
// grid = (UNITS/16 = 128, BATCH/256 = 32), block = 256 (8 waves).
// Wave w: rows [blockIdx.y*256 + w*32, +32) (two 16-row tiles),
// cols [blockIdx.x*16, +16). 8 f32 accumulators.
__global__ __launch_bounds__(256)
void gru_fused(const __bf16* __restrict__ xb, const __bf16* __restrict__ hb,
               const __bf16* __restrict__ wib, const __bf16* __restrict__ whb,
               const float* __restrict__ h_prev, const float* __restrict__ bias,
               float* __restrict__ out) {
    const int lane = threadIdx.x & 31;
    const int wv   = threadIdx.x >> 5;
    const int m0   = blockIdx.y * 256 + wv * 32;
    const int ntu  = blockIdx.x;                 // unit column tile, 0..127
    const int col  = ntu * 16 + (lane & 15);

    v8f r0 = {}, u0 = {}, c0 = {}, h0 = {};
    v8f r1 = {}, u1 = {}, c1 = {}, h1 = {};

    // A: lane holds row M = lane&15; K = kbase..+15 contiguous,
    // kbase = (lane>>4)*16. Second M-tile reached by +16*KDIM immediate.
    const int arow  = m0 + (lane & 15);
    const int kbase = (lane >> 4) * 16;
    const size_t laneOff = (size_t)lane * 16;

    Ptrs p;
    p.xA = xb + (size_t)arow * KDIM + kbase;
    p.hA = hb + (size_t)arow * KDIM + kbase;
    p.wi = wib + (size_t)ntu * (NT_K * 3 * 512) + laneOff;
    p.wh = whb + (size_t)ntu * (NT_K * 3 * 512) + laneOff;

    Frag10 fa = load_all(p, 0);
    Frag10 fb;

    // 15 groups of 4 tiles (tiles 0..59), pointers advance 4 tiles/iter.
#pragma unroll 1
    for (int g = 0; g < 15; ++g) {
        fb = load_all(p, 1); mma_all(fa, r0,u0,c0,h0, r1,u1,c1,h1);
        fa = load_all(p, 2); mma_all(fb, r0,u0,c0,h0, r1,u1,c1,h1);
        fb = load_all(p, 3); mma_all(fa, r0,u0,c0,h0, r1,u1,c1,h1);
        fa = load_all(p, 4); mma_all(fb, r0,u0,c0,h0, r1,u1,c1,h1);
        p.xA += 4 * 32;  p.hA += 4 * 32;
        p.wi += 4 * 3 * 512;  p.wh += 4 * 3 * 512;
    }
    // Tail: fa holds tile 60; tiles 61..63 remain.
    fb = load_all(p, 1); mma_all(fa, r0,u0,c0,h0, r1,u1,c1,h1);
    fa = load_all(p, 2); mma_all(fb, r0,u0,c0,h0, r1,u1,c1,h1);
    fb = load_all(p, 3); mma_all(fa, r0,u0,c0,h0, r1,u1,c1,h1);
    mma_all(fb, r0,u0,c0,h0, r1,u1,c1,h1);

    // GRU epilogue. C/D layout: VGPR r, lane l -> row m0 + r + 8*(l>>4) for
    // tile 0, +16 rows for tile 1; col = ntu*16 + (l&15).
    const float brv = bias[col];
    const float buv = bias[col + UNITS];
    const float bhv = bias[col + 2 * UNITS];
    const int rbase = m0 + ((lane >> 4) * 8);

#pragma unroll
    for (int r = 0; r < 8; ++r) {
        {   // M-tile 0
            const int row = rbase + r;
            const float hp = h_prev[(size_t)row * UNITS + col];
            const float rt = 1.0f / (1.0f + __expf(-(r0[r] + brv)));
            const float ut = 1.0f / (1.0f + __expf(-(u0[r] + buv)));
            const float cand = tanhf(c0[r] + rt * h0[r] + bhv);
            out[(size_t)row * UNITS + col] = ut * hp + (1.0f - ut) * cand;
        }
        {   // M-tile 1 (+16 rows)
            const int row = rbase + 16 + r;
            const float hp = h_prev[(size_t)row * UNITS + col];
            const float rt = 1.0f / (1.0f + __expf(-(r1[r] + brv)));
            const float ut = 1.0f / (1.0f + __expf(-(u1[r] + buv)));
            const float cand = tanhf(c1[r] + rt * h1[r] + bhv);
            out[(size_t)row * UNITS + col] = ut * hp + (1.0f - ut) * cand;
        }
    }
}

// ---------------------------------------------------------------------------
extern "C" void kernel_launch(void* const* d_in, const int* in_sizes, int n_in,
                              void* d_out, int out_size, void* d_ws, size_t ws_size,
                              hipStream_t stream) {
    const float* x    = (const float*)d_in[0];   // [8192, 2048]
    const float* h    = (const float*)d_in[1];   // [8192, 2048]
    const float* Wi   = (const float*)d_in[2];   // [2048, 6144]
    const float* Wh   = (const float*)d_in[3];   // [2048, 6144]
    const float* bias = (const float*)d_in[4];   // [6144]
    float* out = (float*)d_out;

    char* ws = (char*)d_ws;
    __bf16* xb  = (__bf16*)(ws);                        // 32 MiB
    __bf16* hb  = (__bf16*)(ws + (size_t)33554432);     // 32 MiB
    __bf16* wib = (__bf16*)(ws + (size_t)67108864);     // 24 MiB
    __bf16* whb = (__bf16*)(ws + (size_t)92274688);     // 24 MiB

    const int nA = BATCH * KDIM;                        // 16.78M elems
    conv_a_bf16<<<nA / 8 / 256, 256, 0, stream>>>(x, xb, nA);
    conv_a_bf16<<<nA / 8 / 256, 256, 0, stream>>>(h, hb, nA);

    const int jobs = NT_N * NT_K * 32;                  // 786432 tile-lanes
    pack_w_bf16<<<jobs / 256, 256, 0, stream>>>(Wi, wib);
    pack_w_bf16<<<jobs / 256, 256, 0, stream>>>(Wh, whb);

    dim3 grid(UNITS / 16, BATCH / 256);                 // (128, 32)
    gru_fused<<<grid, 256, 0, stream>>>(xb, hb, wib, whb, h, bias, out);
}
